// HeteroSTBlock_66065186947524
// MI455X (gfx1250) — matrix operand order repaired and checked by reading, not compiled, not run on hardware
//
#include <hip/hip_runtime.h>
#include <hip/hip_bf16.h>

// ---------------------------------------------------------------------------
// MI455X (gfx1250) implementation of the HeteroSTBlock reference.
//   * GLU temporal convs  -> implicit GEMM on v_wmma_f32_16x16x32_bf16
//                            (4 position-tiles per block to amortize the
//                            128x192 weight LDS staging 4x)
//   * GCN/SAGE x@W        -> 64x64 WMMA GEMMs over flat (t,node,c) features
//                            (8 row-tiles per block + global_prefetch_b8)
//   * edge scatter/gather -> global_atomic_add_f32 (bandwidth bound)
//   * LayerNorm fused read of flat rows, writes final NCHW output
// Workspace requirement: ~700 MB (documented carve below).
// ---------------------------------------------------------------------------

typedef __bf16 bf16_t;
typedef bf16_t v16bf __attribute__((ext_vector_type(16)));
typedef bf16_t v8bf  __attribute__((ext_vector_type(8)));
typedef float  v8f   __attribute__((ext_vector_type(8)));

static constexpr int BSZ  = 16;    // batch
static constexpr int T1   = 14;
static constexpr int T2   = 12;
static constexpr int NR   = 256, NDV = 1024, NP = 4096;
static constexpr long NN_R = (long)BSZ * NR;    // 4096
static constexpr long NN_D = (long)BSZ * NDV;   // 16384
static constexpr long NN_P = (long)BSZ * NP;    // 65536
static constexpr int CONV_TILES = 4;   // 16-position tiles per conv block
static constexpr int GEMM_RITER = 8;   // 64-row tiles per gemm block

// ----------------------------- WMMA fragment helpers -----------------------
// A-matrix (16x32 bf16): lane l holds row M=l&15; k = kc*32 + (l>>4)*8 + i for
// i<8 and +16 for the upper 8 elements (ISA 7.12.2).
__device__ __forceinline__ v16bf frag_a_lds(const bf16_t* row, int koff) {
  v8bf lo = *(const v8bf*)(row + koff);
  v8bf hi = *(const v8bf*)(row + koff + 16);
  v16bf r;
#pragma unroll
  for (int i = 0; i < 8; ++i) { r[i] = lo[i]; r[i + 8] = hi[i]; }
  return r;
}
// B-matrix (32x16 bf16): lane l holds col N=l&15; k = kc*32 + (l>>4)*8 + i for
// i<8 and +16 for the upper 8 elements.
__device__ __forceinline__ v16bf frag_b_lds(const bf16_t* col, int koff) {
  v8bf lo = *(const v8bf*)(col + koff);
  v8bf hi = *(const v8bf*)(col + koff + 16);
  v16bf r;
#pragma unroll
  for (int i = 0; i < 8; ++i) { r[i] = lo[i]; r[i + 8] = hi[i]; }
  return r;
}
template <typename T>
__device__ __forceinline__ v16bf frag_a_glb(const T* row, int koff) {
  v16bf r;
#pragma unroll
  for (int i = 0; i < 8; ++i) {
    r[i]     = (bf16_t)(float)row[koff + i];
    r[i + 8] = (bf16_t)(float)row[koff + 16 + i];
  }
  return r;
}
__device__ __forceinline__ v8f wmma_bf(v16bf a, v16bf b, v8f c) {
  return __builtin_amdgcn_wmma_f32_16x16x32_bf16(false, a, false, b,
                                                 (short)0, c, false, false);
}

// ----------------------------- GLU conv (implicit GEMM) --------------------
// D = W(128x192) x Xtile(192x16).  Block = 128 threads = 4 waves; wave w owns
// P-tile (oc 16w..) and Q-tile (oc 64+16w..) for the same 16 positions, so the
// GLU epilogue (P + x_in)*sigmoid(Q) is wave-local.  The block iterates over
// CONV_TILES consecutive 16-position tiles, re-staging only the 6KB X-tile and
// reusing the 48KB bf16 weight tile in LDS.  Output written in flat
// (t, b*Nn+n, c) layout (contiguous channels) for downstream WMMA GEMMs.
template <typename InT, typename OutT>
__global__ void __launch_bounds__(128)
glu_conv_wmma(const InT* __restrict__ x, const float* __restrict__ W,
              const float* __restrict__ bias, OutT* __restrict__ out,
              int Nn, int T_out, long sB, long sC, long sT, long sN) {
  __shared__ __align__(16) bf16_t Wlds[128 * 192];
  __shared__ __align__(16) bf16_t Xlds[16 * 192];
  const int tid = threadIdx.x;
  for (int e = tid; e < 128 * 192; e += 128) Wlds[e] = (bf16_t)W[e];

  const int wave = tid >> 5, lane = tid & 31;
  const int col = lane & 15, half = lane >> 4;
  const int ocP = wave * 16, ocQ = 64 + wave * 16;

  for (int it = 0; it < CONV_TILES; ++it) {
    const long p0 = (long)blockIdx.x * (16 * CONV_TILES) + it * 16;
    const int  n0 = (int)(p0 % Nn);        // Nn % 64 == 0 -> fixed (b,t) per tile
    const long q  = p0 / Nn;
    const int  t  = (int)(q % T_out);
    const int  b  = (int)(q / T_out);

    if (it) __syncthreads();               // previous tile's Xlds readers done
    for (int e = tid; e < 192 * 16; e += 128) {  // Xtile[pos][k], k = ic*3+kk
      const int cc = e & 15, r = e >> 4;
      const int ic = r / 3, kk = r - ic * 3;
      const float v = (float)x[(long)b * sB + (long)ic * sC +
                               (long)(t + kk) * sT + (long)(n0 + cc) * sN];
      Xlds[cc * 192 + r] = (bf16_t)v;
    }
    __syncthreads();

    v8f accP = {}, accQ = {};
#pragma unroll
    for (int kc = 0; kc < 6; ++kc) {       // K = 192 = 6 x 32
      v16bf bf = frag_b_lds(&Xlds[col * 192], kc * 32 + half * 8);
      v16bf aP = frag_a_lds(&Wlds[(ocP + col) * 192], kc * 32 + half * 8);
      accP = wmma_bf(aP, bf, accP);
      v16bf aQ = frag_a_lds(&Wlds[(ocQ + col) * 192], kc * 32 + half * 8);
      accQ = wmma_bf(aQ, bf, accQ);
    }
    // D layout: lane holds N=col, M = r + 8*half  ->  oc = ocP + 8*half + r
    const long node = (long)b * Nn + (n0 + col);
    const long orow = ((long)t * ((long)BSZ * Nn) + node) * 64;
#pragma unroll
    for (int r = 0; r < 8; ++r) {
      const int oc = ocP + 8 * half + r;
      const float xin = (float)x[(long)b * sB + (long)oc * sC +
                                 (long)(t + 2) * sT + (long)(n0 + col) * sN];
      const float P = accP[r] + bias[oc];
      const float Q = accQ[r] + bias[64 + oc];
      out[orow + oc] = (OutT)((P + xin) * (1.0f / (1.0f + __expf(-Q))));
    }
  }
}

// Scalar GLU conv for the tiny outside/time streams (N==1 positions).
template <typename InT, typename OutT>
__global__ void glu_small(const InT* __restrict__ x, const float* __restrict__ W,
                          const float* __restrict__ bias, OutT* __restrict__ out,
                          int Nn, int T_out, long sB, long sC, long sT, long sN) {
  const int p = blockIdx.x, oc = threadIdx.x;
  const int n = p % Nn, t = (p / Nn) % T_out, b = p / (Nn * T_out);
  float P = bias[oc], Q = bias[64 + oc];
  for (int ic = 0; ic < 64; ++ic)
#pragma unroll
    for (int k = 0; k < 3; ++k) {
      const float xv = (float)x[(long)b * sB + (long)ic * sC +
                                (long)(t + k) * sT + (long)n * sN];
      P += W[(long)oc * 192 + ic * 3 + k] * xv;
      Q += W[(long)(64 + oc) * 192 + ic * 3 + k] * xv;
    }
  const float xin = (float)x[(long)b * sB + (long)oc * sC +
                             (long)(t + 2) * sT + (long)n * sN];
  out[(((long)t * BSZ + b) * Nn + n) * 64 + oc] =
      (OutT)((P + xin) * (1.0f / (1.0f + __expf(-Q))));
}

// ----------------------------- 64x64 WMMA GEMM -----------------------------
// out[rows x 64] += A[rows x 64] @ Wg[64 x 64].  A-fragments straight from
// global (rows are channel-contiguous), Wt staged transposed in LDS once per
// block and reused for GEMM_RITER row-tiles; next tile prefetched via
// global_prefetch_b8.
template <typename AT>
__global__ void __launch_bounds__(128)
gemm64_add(const AT* __restrict__ A, const float* __restrict__ Wg,
           float* __restrict__ out, long rows) {
  __shared__ __align__(16) bf16_t Wt[64 * 64];  // Wt[d][c] = Wg[c][d]
  const int tid = threadIdx.x;
  for (int e = tid; e < 4096; e += 128) {
    const int d = e >> 6, c = e & 63;
    Wt[d * 64 + c] = (bf16_t)Wg[c * 64 + d];
  }
  __syncthreads();
  const int wave = tid >> 5, lane = tid & 31;
  const int col = lane & 15, half = lane >> 4;
  for (int it = 0; it < GEMM_RITER; ++it) {
    const long row0 = ((long)blockIdx.x * GEMM_RITER + it) * 64 + wave * 16;
    const AT* arow = A + (row0 + col) * 64;
    if (it + 1 < GEMM_RITER)
      __builtin_prefetch(arow + 64 * 64, 0, 0);   // next row-tile
    v8f acc[4] = {{}, {}, {}, {}};
#pragma unroll
    for (int kc = 0; kc < 2; ++kc) {
      v16bf af = frag_a_glb(arow, kc * 32 + half * 8);
#pragma unroll
      for (int dt = 0; dt < 4; ++dt) {
        v16bf bf = frag_b_lds(&Wt[(dt * 16 + col) * 64], kc * 32 + half * 8);
        acc[dt] = wmma_bf(af, bf, acc[dt]);
      }
    }
#pragma unroll
    for (int dt = 0; dt < 4; ++dt)
#pragma unroll
      for (int r = 0; r < 8; ++r) {
        const long rr = row0 + r + 8 * half;
        out[rr * 64 + dt * 16 + col] += acc[dt][r];
      }
  }
}

// ----------------------------- element-wise / graph kernels ----------------
__global__ void fill_f32(float* p, float v, long n) {
  const long i = (long)blockIdx.x * blockDim.x + threadIdx.x;
  if (i < n) p[i] = v;
}
__global__ void fill_rows64(float* p, const float* bias, long n) {
  const long i = (long)blockIdx.x * blockDim.x + threadIdx.x;
  if (i < n) p[i] = bias[i & 63];
}
__global__ void bias_sums(const float* gb, const float* bl, float* bsum) {
  const int c = threadIdx.x;
  bsum[c]       = gb[c] + gb[64 + c] + bl[c] + bl[128 + c] + bl[192 + c];
  bsum[64 + c]  = bl[64 + c] + bl[256 + c];
  bsum[128 + c] = bl[320 + c];
}
__global__ void flatten_bf(const float* __restrict__ x, bf16_t* __restrict__ y,
                           int Nsec, long total) {
  const long i = (long)blockIdx.x * blockDim.x + threadIdx.x;
  if (i >= total) return;
  const int c = (int)(i & 63);
  const long rest = i >> 6;
  const long NN = (long)BSZ * Nsec;
  const int t = (int)(rest / NN);
  const long node = rest % NN;
  const int b = (int)(node / Nsec), n = (int)(node % Nsec);
  y[i] = (bf16_t)x[(((long)b * 64 + c) * 16 + t) * Nsec + n];
}
__global__ void scatter_w(const int* __restrict__ d, const float* __restrict__ w,
                          int E, float* acc) {
  const int e = blockIdx.x * blockDim.x + threadIdx.x;
  if (e < E) atomicAdd(&acc[d[e]], w ? w[e] : 1.0f);
}
__global__ void add_one(float* p, int n) {
  const int i = blockIdx.x * blockDim.x + threadIdx.x;
  if (i < n) p[i] += 1.0f;
}
__global__ void dinv_k(float* p, int n) {
  const int i = blockIdx.x * blockDim.x + threadIdx.x;
  if (i < n) { const float d = p[i]; p[i] = d > 0.f ? rsqrtf(d) : 0.f; }
}
__global__ void norm_k(const int* s, const int* d, const float* w,
                       const float* dinv, float* nrm, int E) {
  const int e = blockIdx.x * blockDim.x + threadIdx.x;
  if (e < E) nrm[e] = dinv[s[e]] * w[e] * dinv[d[e]];
}
template <typename FT>
__global__ void scatter_feat(const FT* __restrict__ feat, const int* __restrict__ src,
                             const int* __restrict__ dst, int E, int n_src, int n_dst,
                             const float* __restrict__ scale, float* __restrict__ agg) {
  const long total = (long)E * T1 * 64;
  const long i = (long)blockIdx.x * blockDim.x + threadIdx.x;
  if (i >= total) return;
  const int c = (int)(i & 63);
  const long rest = i >> 6;
  const int t = (int)(rest % T1);
  const int e = (int)(rest / T1);
  float v = (float)feat[((long)t * n_src + src[e]) * 64 + c];
  if (scale) v *= scale[e];
  atomicAdd(&agg[((long)t * n_dst + dst[e]) * 64 + c], v);
}
__global__ void gcn_self(const float* __restrict__ h, const float* __restrict__ dinv,
                         float* __restrict__ outs, int n, long total) {
  const long i = (long)blockIdx.x * blockDim.x + threadIdx.x;
  if (i >= total) return;
  const int node = (int)((i >> 6) % n);
  const float dv = dinv[node];
  outs[i] += h[i] * dv * dv;
}
__global__ void mean_div(float* agg, const float* cnt, int n_dst, long total) {
  const long i = (long)blockIdx.x * blockDim.x + threadIdx.x;
  if (i >= total) return;
  agg[i] /= fmaxf(cnt[(int)((i >> 6) % n_dst)], 1.0f);
}
template <typename T>
__global__ void relu_to_bf(const T* __restrict__ x, bf16_t* __restrict__ y, long n) {
  const long i = (long)blockIdx.x * blockDim.x + threadIdx.x;
  if (i < n) y[i] = (bf16_t)fmaxf((float)x[i], 0.0f);
}
// LayerNorm over channels of a flat row, writes final (b,c,t,n) output.
__global__ void __launch_bounds__(128)
ln_store(const float* __restrict__ xf, long NN, int Nsec, int doRelu,
         const float* __restrict__ g, const float* __restrict__ bta,
         float* __restrict__ outp) {
  const long rows = (long)T2 * NN;
  const long i = (long)blockIdx.x * blockDim.x + threadIdx.x;
  if (i >= rows) return;
  const int t = (int)(i / NN);
  const long node = i % NN;
  const float* r = xf + ((long)t * NN + node) * 64;
  float s = 0.f, ss = 0.f;
  for (int c = 0; c < 64; ++c) {
    float v = r[c];
    if (doRelu) v = fmaxf(v, 0.f);
    s += v; ss += v * v;
  }
  const float m = s * (1.0f / 64.0f);
  const float var = fmaxf(ss * (1.0f / 64.0f) - m * m, 0.0f);
  const float inv = rsqrtf(var + 1e-5f);
  const int b = (int)(node / Nsec), n = (int)(node % Nsec);
  for (int c = 0; c < 64; ++c) {
    float v = r[c];
    if (doRelu) v = fmaxf(v, 0.f);
    outp[(((long)b * 64 + c) * T2 + t) * Nsec + n] = (v - m) * inv * g[c] + bta[c];
  }
}

// ----------------------------- host orchestration --------------------------
extern "C" void kernel_launch(void* const* d_in, const int* in_sizes, int n_in,
                              void* d_out, int out_size, void* d_ws, size_t ws_size,
                              hipStream_t stream) {
  (void)in_sizes; (void)n_in; (void)out_size; (void)ws_size;
  const float* x_room  = (const float*)d_in[0];
  const float* x_dev   = (const float*)d_in[1];
  const float* x_prop  = (const float*)d_in[2];
  const float* x_out   = (const float*)d_in[3];
  const float* x_time  = (const float*)d_in[4];
  const float* c1W = (const float*)d_in[5];
  const float* c1b = (const float*)d_in[6];
  const float* c2W = (const float*)d_in[7];
  const float* c2b = (const float*)d_in[8];
  const float* gcnW = (const float*)d_in[9];
  const float* gcnB = (const float*)d_in[10];
  const float* sWl = (const float*)d_in[11];
  const float* sBl = (const float*)d_in[12];
  const float* sWr = (const float*)d_in[13];
  const float* lnG = (const float*)d_in[14];
  const float* lnB = (const float*)d_in[15];
  const float* hh_w = (const float*)d_in[16];
  const float* hv_w = (const float*)d_in[17];
  const int *hh_s = (const int*)d_in[18], *hh_d = (const int*)d_in[19];
  const int *hv_s = (const int*)d_in[20], *hv_d = (const int*)d_in[21];
  const int *dr_s = (const int*)d_in[22], *dr_d = (const int*)d_in[23];
  const int *pd_s = (const int*)d_in[24], *pd_d = (const int*)d_in[25];
  const int *or_s = (const int*)d_in[26], *or_d = (const int*)d_in[27];
  const int *tr_s = (const int*)d_in[28], *tr_d = (const int*)d_in[29];
  const int *td_s = (const int*)d_in[30], *td_d = (const int*)d_in[31];
  const int *tp_s = (const int*)d_in[32], *tp_d = (const int*)d_in[33];
  float* outp = (float*)d_out;

  // ---- workspace carve (~700 MB) ----
  size_t off = 0;
  auto carve = [&](size_t bytes) -> void* {
    off = (off + 255) & ~(size_t)255;
    void* p = (char*)d_ws + off;
    off += bytes;
    return p;
  };
  bf16_t* rbf  = (bf16_t*)carve((size_t)T1 * NN_R * 64 * 2);   //   7.3 MB
  bf16_t* dbf  = (bf16_t*)carve((size_t)T1 * NN_D * 64 * 2);   //  29.4 MB
  bf16_t* pbf  = (bf16_t*)carve((size_t)T1 * NN_P * 64 * 2);   // 117.4 MB
  bf16_t* obf  = (bf16_t*)carve((size_t)T1 * 16 * 64 * 2);
  bf16_t* tbf  = (bf16_t*)carve((size_t)T1 * 16 * 64 * 2);
  bf16_t* obf2 = (bf16_t*)carve((size_t)T1 * 16 * 64 * 2);
  bf16_t* tbf2 = (bf16_t*)carve((size_t)T1 * 16 * 64 * 2);
  float* room_s = (float*)carve((size_t)T1 * NN_R * 64 * 4);   //  14.7 MB
  float* dev_s  = (float*)carve((size_t)T1 * NN_D * 64 * 4);   //  58.7 MB
  float* prop_s = (float*)carve((size_t)T1 * NN_P * 64 * 4);   // 234.9 MB
  float* agg    = (float*)carve((size_t)T1 * NN_P * 64 * 4);   // 234.9 MB (reused)
  float* cnt    = (float*)carve((size_t)NN_P * 4);
  float* nrm    = (float*)carve((size_t)16384 * 4);
  float* bsum   = (float*)carve(3 * 64 * 4);
  float* o3f    = (float*)carve((size_t)T2 * 16 * 64 * 4);
  float* t3f    = (float*)carve((size_t)T2 * 16 * 64 * 4);
  bf16_t* p2bf  = (bf16_t*)agg;     // prop2 (bf16) overlays agg after SAGE done
  float*  prop3 = prop_s;           // conv2 output overlays prop_s after relu

  auto nb = [](long n) { return (unsigned)((n + 255) / 256); };

  const long fR = (long)T1 * NN_R * 64, fD = (long)T1 * NN_D * 64, fP = (long)T1 * NN_P * 64;

  // biases summed per destination node-type, broadcast into the accumulators
  bias_sums<<<1, 64, 0, stream>>>(gcnB, sBl, bsum);
  fill_rows64<<<nb(fR), 256, 0, stream>>>(room_s, bsum, fR);
  fill_rows64<<<nb(fD), 256, 0, stream>>>(dev_s, bsum + 64, fD);
  fill_rows64<<<nb(fP), 256, 0, stream>>>(prop_s, bsum + 128, fP);

  // flat bf16 features of the sliced room/device streams
  flatten_bf<<<nb(fR), 256, 0, stream>>>(x_room, rbf, NR, fR);
  flatten_bf<<<nb(fD), 256, 0, stream>>>(x_dev, dbf, NDV, fD);

  // stage-1 GLU convs: property via WMMA implicit GEMM; outside/time scalar.
  glu_conv_wmma<float, bf16_t>
      <<<(unsigned)(BSZ * T1 * NP / (16 * CONV_TILES)), 128, 0, stream>>>(
      x_prop, c1W, c1b, pbf, NP, T1,
      (long)64 * 16 * NP, (long)16 * NP, (long)NP, 1L);
  glu_small<float, bf16_t><<<BSZ * T1, 64, 0, stream>>>(
      x_out, c1W + 24576, c1b + 128, obf, 1, T1, 1024L, 16L, 1L, 1L);
  glu_small<float, bf16_t><<<BSZ * T1, 64, 0, stream>>>(
      x_time, c1W + 49152, c1b + 256, tbf, 1, T1, 1024L, 16L, 1L, 1L);

  // ---- GCNs on room features ----
  auto gcn = [&](const int* si, const int* di, const float* wi, int E, const float* Wg) {
    fill_f32<<<nb(NN_R), 256, 0, stream>>>(cnt, 0.f, NN_R);
    scatter_w<<<nb(E), 256, 0, stream>>>(di, wi, E, cnt);
    add_one<<<nb(NN_R), 256, 0, stream>>>(cnt, (int)NN_R);
    dinv_k<<<nb(NN_R), 256, 0, stream>>>(cnt, (int)NN_R);     // cnt = dinv
    norm_k<<<nb(E), 256, 0, stream>>>(si, di, wi, cnt, nrm, E);
    fill_f32<<<nb(fR), 256, 0, stream>>>(agg, 0.f, fR);       // agg = h
    gemm64_add<bf16_t>
        <<<(unsigned)((T1 * NN_R) / (64 * GEMM_RITER)), 128, 0, stream>>>(
        rbf, Wg, agg, T1 * NN_R);
    scatter_feat<float><<<nb((long)E * T1 * 64), 256, 0, stream>>>(
        agg, si, di, E, (int)NN_R, (int)NN_R, nrm, room_s);
    gcn_self<<<nb(fR), 256, 0, stream>>>(agg, cnt, room_s, (int)NN_R, fR);
  };
  gcn(hh_s, hh_d, hh_w, 16384, gcnW);
  gcn(hv_s, hv_d, hv_w, 8192, gcnW + 4096);

  // ---- SAGE layers (sequentially reusing `agg`) ----
  auto sage = [&](const bf16_t* srcf, int n_src, const bf16_t* dstf, long n_dst,
                  const int* si, const int* di, int E,
                  const float* Wl, const float* Wr, float* outs) {
    const long an = (long)T1 * n_dst * 64;
    fill_f32<<<nb(an), 256, 0, stream>>>(agg, 0.f, an);
    fill_f32<<<nb(n_dst), 256, 0, stream>>>(cnt, 0.f, n_dst);
    scatter_w<<<nb(E), 256, 0, stream>>>(di, (const float*)nullptr, E, cnt);
    scatter_feat<bf16_t><<<nb((long)E * T1 * 64), 256, 0, stream>>>(
        srcf, si, di, E, n_src, (int)n_dst, nullptr, agg);
    mean_div<<<nb(an), 256, 0, stream>>>(agg, cnt, (int)n_dst, an);
    gemm64_add<float>
        <<<(unsigned)((T1 * n_dst) / (64 * GEMM_RITER)), 128, 0, stream>>>(
        agg, Wl, outs, T1 * n_dst);
    gemm64_add<bf16_t>
        <<<(unsigned)((T1 * n_dst) / (64 * GEMM_RITER)), 128, 0, stream>>>(
        dstf, Wr, outs, T1 * n_dst);
  };
  sage(dbf, (int)NN_D, rbf, NN_R, dr_s, dr_d, 16384, sWl + 0 * 4096, sWr + 0 * 4096, room_s);
  sage(obf, 16, rbf, NN_R, or_s, or_d, 4096, sWl + 2 * 4096, sWr + 2 * 4096, room_s);
  sage(tbf, 16, rbf, NN_R, tr_s, tr_d, 4096, sWl + 3 * 4096, sWr + 3 * 4096, room_s);
  sage(pbf, (int)NN_P, dbf, NN_D, pd_s, pd_d, 65536, sWl + 1 * 4096, sWr + 1 * 4096, dev_s);
  sage(tbf, 16, dbf, NN_D, td_s, td_d, 16384, sWl + 4 * 4096, sWr + 4 * 4096, dev_s);
  sage(tbf, 16, pbf, NN_P, tp_s, tp_d, 65536, sWl + 5 * 4096, sWr + 5 * 4096, prop_s);

  // ---- ReLU + stage-2 GLU convs ----
  relu_to_bf<float><<<nb(fP), 256, 0, stream>>>(prop_s, p2bf, fP);
  relu_to_bf<bf16_t><<<nb((long)T1 * 16 * 64), 256, 0, stream>>>(obf, obf2, (long)T1 * 16 * 64);
  relu_to_bf<bf16_t><<<nb((long)T1 * 16 * 64), 256, 0, stream>>>(tbf, tbf2, (long)T1 * 16 * 64);

  // prop2 flat strides as (b, ic, t, n):  idx = ((t*NN_P + b*NP + n)*64 + ic)
  glu_conv_wmma<bf16_t, float>
      <<<(unsigned)(BSZ * T2 * NP / (16 * CONV_TILES)), 128, 0, stream>>>(
      p2bf, c2W, c2b, prop3, NP, T2,
      (long)NP * 64, 1L, (long)NN_P * 64, 64L);
  glu_small<bf16_t, float><<<BSZ * T2, 64, 0, stream>>>(
      obf2, c2W + 24576, c2b + 128, o3f, 1, T2, 64L, 1L, 1024L, 64L);
  glu_small<bf16_t, float><<<BSZ * T2, 64, 0, stream>>>(
      tbf2, c2W + 49152, c2b + 256, t3f, 1, T2, 64L, 1L, 1024L, 64L);

  // ---- LayerNorm + final NCHW store (section offsets in return order) ----
  long o = 0;
  ln_store<<<nb(T2 * NN_R), 128, 0, stream>>>(room_s, NN_R, NR, 1, lnG + 0, lnB + 0, outp + o);
  o += (long)BSZ * 64 * T2 * NR;
  ln_store<<<nb(T2 * NN_D), 128, 0, stream>>>(dev_s, NN_D, NDV, 1, lnG + 64, lnB + 64, outp + o);
  o += (long)BSZ * 64 * T2 * NDV;
  ln_store<<<nb(T2 * NN_P), 128, 0, stream>>>(prop3, NN_P, NP, 0, lnG + 128, lnB + 128, outp + o);
  o += (long)BSZ * 64 * T2 * NP;
  ln_store<<<nb((long)T2 * 16), 128, 0, stream>>>(o3f, 16, 1, 0, lnG + 192, lnB + 192, outp + o);
  o += (long)BSZ * 64 * T2;
  ln_store<<<nb((long)T2 * 16), 128, 0, stream>>>(t3f, 16, 1, 0, lnG + 256, lnB + 256, outp + o);
}
